// _RNN_wrapper_6605659701442
// MI455X (gfx1250) — matrix operand order repaired and checked by reading, not compiled, not run on hardware
//
#include <hip/hip_runtime.h>
#include <hip/hip_bf16.h>
#include <math.h>

// Problem constants (fixed by the reference)
#define BB        16
#define NTUBES    64
#define TT        8
#define DK        25088          // K dimension
#define CC        1000           // real classes
#define CPAD      1024           // padded N for full 16x16 tiles
#define MROWS     (BB * NTUBES)  // 1024

typedef __bf16 bf16_t;
typedef __attribute__((ext_vector_type(4)))  __bf16 v4bf;
typedef __attribute__((ext_vector_type(8)))  __bf16 v8bf;
typedef __attribute__((ext_vector_type(16))) __bf16 v16bf;
typedef __attribute__((ext_vector_type(8)))  float  v8f;

// ---------------------------------------------------------------------------
// Kernel 1: masked temporal mean-pool, f32 -> bf16.
// pooled[row, d] = (1/len_b) * sum_{t<len_b} features[b, n, t, d]
// Streaming, memory-bound: only reads the t < len rows (saves ~40% of 822MB).
// ---------------------------------------------------------------------------
__global__ __launch_bounds__(256)
void pool_to_bf16_kernel(const float* __restrict__ feats,
                         const int*   __restrict__ len_tubes,
                         bf16_t*      __restrict__ pooled) {
    const int row = blockIdx.y;                              // 0..1023 = b*64+n
    const int c4  = blockIdx.x * blockDim.x + threadIdx.x;   // float4 column
    if (c4 >= DK / 4) return;
    const int b   = row >> 6;
    int len = len_tubes[b];
    len = len < 1 ? 1 : (len > TT ? TT : len);

    const float* base = feats + (size_t)row * TT * DK + (size_t)c4 * 4;
    float ax = 0.f, ay = 0.f, az = 0.f, aw = 0.f;
    for (int t = 0; t < len; ++t) {
        const float4 v = *(const float4*)(base + (size_t)t * DK);
        ax += v.x; ay += v.y; az += v.z; aw += v.w;
    }
    const float inv = 1.0f / (float)len;
    v4bf o;
    o[0] = (bf16_t)(ax * inv);
    o[1] = (bf16_t)(ay * inv);
    o[2] = (bf16_t)(az * inv);
    o[3] = (bf16_t)(aw * inv);
    *(v4bf*)(pooled + (size_t)row * DK + (size_t)c4 * 4) = o;
}

// ---------------------------------------------------------------------------
// Kernel 2: W (f32, [C][D]) -> bf16 [CPAD][D], rows >= C zero-filled so the
// padded GEMM columns produce exact zeros (ignored downstream anyway).
// ---------------------------------------------------------------------------
__global__ __launch_bounds__(256)
void w_to_bf16_kernel(const float* __restrict__ W,
                      bf16_t*      __restrict__ Wb) {
    const int c  = blockIdx.y;                               // 0..1023
    const int c4 = blockIdx.x * blockDim.x + threadIdx.x;
    if (c4 >= DK / 4) return;
    v4bf o;
    if (c < CC) {
        const float4 v = *(const float4*)(W + (size_t)c * DK + (size_t)c4 * 4);
        o[0] = (bf16_t)v.x; o[1] = (bf16_t)v.y;
        o[2] = (bf16_t)v.z; o[3] = (bf16_t)v.w;
    } else {
        o[0] = (bf16_t)0.f; o[1] = (bf16_t)0.f;
        o[2] = (bf16_t)0.f; o[3] = (bf16_t)0.f;
    }
    *(v4bf*)(Wb + (size_t)c * DK + (size_t)c4 * 4) = o;
}

// ---------------------------------------------------------------------------
// Kernel 3: bf16 WMMA GEMM.  logits[m,n] = sum_k pooled[m,k] * Wb[n,k]
// M = 1024, N = CPAD = 1024, K = 25088.
// Block tile 128x128, BK = 32, 256 threads = 8 waves, wave tile 64x32
// (4x2 sub-tiles of 16x16). LDS double-buffered, one barrier per K-step;
// global loads for step k+1 issue before the 8 WMMAs of step k.
// No divergence: EXEC is all-ones for every v_wmma.
// ---------------------------------------------------------------------------
#define BM   128
#define BN   128
#define BK   32
#define KP   40   // padded LDS row length (bf16 elements) to dodge bank conflicts

__global__ __launch_bounds__(256)
void gemm_bf16_wmma_kernel(const bf16_t* __restrict__ A,   // [MROWS][DK]
                           const bf16_t* __restrict__ Bw,  // [CPAD][DK]
                           float*        __restrict__ Cout)// [MROWS][CPAD]
{
    __shared__ bf16_t As[2][BM][KP];
    __shared__ bf16_t Bs[2][BN][KP];

    const int tid  = threadIdx.x;
    const int m0   = blockIdx.y * BM;
    const int n0   = blockIdx.x * BN;

    // global->LDS staging pattern: 4 threads per row, 8 bf16 (16B) each,
    // each thread covers rows lrow and lrow+64.
    const int lrow = tid >> 2;            // 0..63
    const int lcol = (tid & 3) * 8;       // 0,8,16,24

    const int lane = tid & 31;
    const int wave = tid >> 5;            // 0..7
    const int wm   = (wave >> 2) * 64;    // 0 | 64
    const int wn   = (wave & 3) * 32;     // 0,32,64,96
    const int half = lane >> 4;           // 0 | 1
    const int l16  = lane & 15;

    const bf16_t* Ag0 = A  + (size_t)(m0 + lrow)      * DK + lcol;
    const bf16_t* Ag1 = A  + (size_t)(m0 + lrow + 64) * DK + lcol;
    const bf16_t* Bg0 = Bw + (size_t)(n0 + lrow)      * DK + lcol;
    const bf16_t* Bg1 = Bw + (size_t)(n0 + lrow + 64) * DK + lcol;

    v8f acc[4][2] = {};

    // preload chunk 0 into LDS buffer 0
    {
        *(v8bf*)&As[0][lrow     ][lcol] = *(const v8bf*)Ag0;
        *(v8bf*)&As[0][lrow + 64][lcol] = *(const v8bf*)Ag1;
        *(v8bf*)&Bs[0][lrow     ][lcol] = *(const v8bf*)Bg0;
        *(v8bf*)&Bs[0][lrow + 64][lcol] = *(const v8bf*)Bg1;
    }
    __syncthreads();

    const int NK = DK / BK;               // 784
    for (int kc = 0; kc < NK; ++kc) {
        const int cur = kc & 1;
        const int nxt = cur ^ 1;

        // issue next chunk's global loads first (latency hidden by WMMAs)
        v8bf ra0, ra1, rb0, rb1;
        const bool more = (kc + 1) < NK;
        if (more) {
            const int ko = (kc + 1) * BK;
            ra0 = *(const v8bf*)(Ag0 + ko);
            ra1 = *(const v8bf*)(Ag1 + ko);
            rb0 = *(const v8bf*)(Bg0 + ko);
            rb1 = *(const v8bf*)(Bg1 + ko);
        }

        // B fragments: lane holds column n=l16, K = half*8..+8 and 16+half*8..+8
        v16bf bfrag[2];
#pragma unroll
        for (int nt = 0; nt < 2; ++nt) {
            const bf16_t* p = &Bs[cur][wn + nt * 16 + l16][half * 8];
            v8bf blo = *(const v8bf*)p;
            v8bf bhi = *(const v8bf*)(p + 16);
            bfrag[nt] = __builtin_shufflevector(blo, bhi,
                0,1,2,3,4,5,6,7,8,9,10,11,12,13,14,15);
        }

#pragma unroll
        for (int mt = 0; mt < 4; ++mt) {
            // A fragment: lane holds row m=l16, K = half*8..+8 and 16+half*8..+8
            const bf16_t* pa = &As[cur][wm + mt * 16 + l16][half * 8];
            v8bf alo = *(const v8bf*)pa;
            v8bf ahi = *(const v8bf*)(pa + 16);
            v16bf afrag = __builtin_shufflevector(alo, ahi,
                0,1,2,3,4,5,6,7,8,9,10,11,12,13,14,15);
#pragma unroll
            for (int nt = 0; nt < 2; ++nt) {
                acc[mt][nt] = __builtin_amdgcn_wmma_f32_16x16x32_bf16(
                    false, afrag, false, bfrag[nt],
                    (short)0, acc[mt][nt], false, false);
            }
        }

        if (more) {
            // buffer `nxt` was last read one barrier ago -> safe to overwrite
            *(v8bf*)&As[nxt][lrow     ][lcol] = ra0;
            *(v8bf*)&As[nxt][lrow + 64][lcol] = ra1;
            *(v8bf*)&Bs[nxt][lrow     ][lcol] = rb0;
            *(v8bf*)&Bs[nxt][lrow + 64][lcol] = rb1;
        }
        __syncthreads();
    }

    // D layout: VGPR r -> M = r + half*8, N = l16
#pragma unroll
    for (int mt = 0; mt < 4; ++mt)
#pragma unroll
        for (int nt = 0; nt < 2; ++nt)
#pragma unroll
            for (int r = 0; r < 8; ++r) {
                const int row = m0 + wm + mt * 16 + half * 8 + r;
                const int col = n0 + wn + nt * 16 + l16;
                Cout[(size_t)row * CPAD + col] = acc[mt][nt][r];
            }
}

// ---------------------------------------------------------------------------
// Kernel 4: per-row log-softmax + NLL (f32 bias added here, only c < 1000).
// One 256-thread block per row; LDS tree reductions (deterministic).
// ---------------------------------------------------------------------------
__global__ __launch_bounds__(256)
void softmax_nll_kernel(const float* __restrict__ logits,
                        const float* __restrict__ bias,
                        const int*   __restrict__ labels,
                        float*       __restrict__ row_nll,
                        float*       __restrict__ row_cnt) {
    __shared__ float red[256];
    const int row = blockIdx.x;
    const int tid = threadIdx.x;
    const float* lr = logits + (size_t)row * CPAD;

    float m = -3.4e38f;
    for (int c = tid; c < CC; c += 256) m = fmaxf(m, lr[c] + bias[c]);
    red[tid] = m; __syncthreads();
    for (int s = 128; s > 0; s >>= 1) {
        if (tid < s) red[tid] = fmaxf(red[tid], red[tid + s]);
        __syncthreads();
    }
    m = red[0]; __syncthreads();

    float sum = 0.f;
    for (int c = tid; c < CC; c += 256) sum += __expf(lr[c] + bias[c] - m);
    red[tid] = sum; __syncthreads();
    for (int s = 128; s > 0; s >>= 1) {
        if (tid < s) red[tid] += red[tid + s];
        __syncthreads();
    }

    if (tid == 0) {
        const float lse = m + __logf(red[0]);
        const int lab = labels[row];
        if (lab >= 0) {
            row_nll[row] = lse - (lr[lab] + bias[lab]);
            row_cnt[row] = 1.0f;
        } else {
            row_nll[row] = 0.0f;
            row_cnt[row] = 0.0f;
        }
    }
}

// ---------------------------------------------------------------------------
// Kernel 5: deterministic final reduction -> scalar loss
// ---------------------------------------------------------------------------
__global__ __launch_bounds__(256)
void finalize_kernel(const float* __restrict__ row_nll,
                     const float* __restrict__ row_cnt,
                     float* __restrict__ out) {
    __shared__ float sn[256], sc[256];
    const int tid = threadIdx.x;
    float a = 0.f, b = 0.f;
    for (int i = tid; i < MROWS; i += 256) { a += row_nll[i]; b += row_cnt[i]; }
    sn[tid] = a; sc[tid] = b; __syncthreads();
    for (int s = 128; s > 0; s >>= 1) {
        if (tid < s) { sn[tid] += sn[tid + s]; sc[tid] += sc[tid + s]; }
        __syncthreads();
    }
    if (tid == 0) out[0] = sn[0] / fmaxf(sc[0], 1.0f);
}

// ---------------------------------------------------------------------------
extern "C" void kernel_launch(void* const* d_in, const int* in_sizes, int n_in,
                              void* d_out, int out_size, void* d_ws, size_t ws_size,
                              hipStream_t stream) {
    const float* feats     = (const float*)d_in[0];  // (16,64,8,25088) f32
    const float* W         = (const float*)d_in[1];  // (1000,25088) f32
    const float* bias      = (const float*)d_in[2];  // (1000,) f32
    const int*   target    = (const int*)d_in[4];    // (16,64) int
    const int*   len_tubes = (const int*)d_in[5];    // (16,1,1) int
    float*       out       = (float*)d_out;

    // workspace layout
    char* ws = (char*)d_ws;
    size_t off = 0;
    bf16_t* pooled = (bf16_t*)(ws + off); off += (size_t)MROWS * DK * sizeof(bf16_t);
    bf16_t* Wb     = (bf16_t*)(ws + off); off += (size_t)CPAD  * DK * sizeof(bf16_t);
    float* logits  = (float*) (ws + off); off += (size_t)MROWS * CPAD * sizeof(float);
    float* row_nll = (float*) (ws + off); off += (size_t)MROWS * sizeof(float);
    float* row_cnt = (float*) (ws + off); off += (size_t)MROWS * sizeof(float);
    (void)ws_size; (void)in_sizes; (void)n_in; (void)out_size;

    // 1) masked mean-pool -> bf16  (dominant memory traffic, ~35us roofline)
    {
        dim3 grid((DK / 4 + 255) / 256, MROWS);
        pool_to_bf16_kernel<<<grid, 256, 0, stream>>>(feats, len_tubes, pooled);
    }
    // 2) W -> bf16 (padded to 1024 rows)
    {
        dim3 grid((DK / 4 + 255) / 256, CPAD);
        w_to_bf16_kernel<<<grid, 256, 0, stream>>>(W, Wb);
    }
    // 3) WMMA GEMM: 1024 x 1024 x 25088
    {
        dim3 grid(CPAD / BN, MROWS / BM);  // 8 x 8
        gemm_bf16_wmma_kernel<<<grid, 256, 0, stream>>>(pooled, Wb, logits);
    }
    // 4) per-row log-softmax + NLL
    softmax_nll_kernel<<<MROWS, 256, 0, stream>>>(logits, bias, target,
                                                  row_nll, row_cnt);
    // 5) scalar loss
    finalize_kernel<<<1, 256, 0, stream>>>(row_nll, row_cnt, out);
}
